// GSABlock_39711267619132
// MI455X (gfx1250) — matrix operand order repaired and checked by reading, not compiled, not run on hardware
//
#include <hip/hip_runtime.h>
#include <hip/hip_bf16.h>
#include <math.h>

typedef __attribute__((ext_vector_type(16))) __bf16 v16bf;
typedef __attribute__((ext_vector_type(8)))  float  v8f;

namespace {
constexpr int Bc = 4;
constexpr int Tc = 2048;
constexpr int Dc = 1024;
constexpr int Hc = 4;
constexpr int KH = 256;          // head dim (K == V)
constexpr int MS = 64;           // slots
constexpr int BT = Bc * Tc;      // 8192 rows
constexpr int HM = Hc * MS;      // 256
constexpr float GATE_INV = 0.125f;   // 1/GATE_LOGIT_NORMALIZER
constexpr float QK_SCALE = 0.0625f;  // K^-0.5
}

// ============================================================================
// Fragment-order packing (ISA 7.12.2, wave32, v_wmma_f32_16x16x32_bf16).
//
// A-operand record layout:  AF[((rt*KC + kc)*32 + lane)*16 + e]
//   rt = row>>4, ar = row&15, kc = K>>5, ko = K&31
//   lane = ar + 16*((ko>>3)&1),  e = (ko&7) + 8*(ko>>4)
// B-operand record layout:  BF[((ct*KC + kc)*32 + lane)*16 + e]
//   ct = n>>4, kc = K>>5, ko = K&31
//   lane = (n&15) + 16*((ko>>3)&1),  e = (ko&7) + 8*(ko>>4)
// Each lane's 16-element fragment is one contiguous, 32B-aligned record.
// ============================================================================

__global__ void pack_a_frag(const float* __restrict__ in, __bf16* __restrict__ out,
                            int rows, int cols) {   // cols % 32 == 0
  const int KC = cols >> 5;
  long long nchunk = (long long)rows * (cols >> 3);  // 8-element chunks
  long long stride = (long long)gridDim.x * blockDim.x;
  for (long long t = (long long)blockIdx.x * blockDim.x + threadIdx.x;
       t < nchunk; t += stride) {
    const int row = (int)(t >> 7);              // cols/8 == 128 for Dc=1024
    const int c8  = ((int)t & 127) * 8;
    const int rt = row >> 4, ar = row & 15;
    const int kc = c8 >> 5, ko = c8 & 31;
    const int lane = ar + ((ko >> 3) & 1) * 16;
    const int half = ko >> 4;
    const float* src = in + (size_t)row * cols + c8;
    __bf16* dst = out + ((((size_t)rt * KC + kc) * 32 + lane) * 16 + half * 8);
#pragma unroll
    for (int e = 0; e < 8; ++e) dst[e] = (__bf16)src[e];
  }
}

__global__ void pack_b_frag(const float* __restrict__ in, __bf16* __restrict__ out,
                            int Kd, int N) {    // in: [Kd][N] row-major
  const int KC = Kd >> 5;
  long long ntot = (long long)Kd * N;
  long long stride = (long long)gridDim.x * blockDim.x;
  for (long long t = (long long)blockIdx.x * blockDim.x + threadIdx.x;
       t < ntot; t += stride) {
    const int kdim = (int)(t / N);
    const int n    = (int)(t % N);
    const int ct = n >> 4, kc = kdim >> 5, ko = kdim & 31;
    const int lane = (n & 15) + ((ko >> 3) & 1) * 16;
    const int e = (ko & 7) + (ko >> 4) * 8;
    out[(((size_t)ct * KC + kc) * 32 + lane) * 16 + e] = (__bf16)in[t];
  }
}

// ============================================================================
// WMMA GEMM, operands pre-swizzled to fragment order: no LDS, no barriers.
// C[Mr x N] fp32 row-major.  MODE 0: none; 1: swish; 2: ge = exp(logsigmoid/8)
// Block = 256x128 (16x8 16x16 tiles); wave = 64x64 (4x4 tiles), 8 waves (4x2).
// Per k-chunk: 16 b128-pair loads feed 16 WMMAs (1:1 issue ratio).
// ============================================================================
template <int MODE>
__global__ __launch_bounds__(256)
void gemm_frag_wmma(const __bf16* __restrict__ AF, const __bf16* __restrict__ BF,
                    float* __restrict__ C, int N, int Kd) {
  const int KC   = Kd >> 5;
  const int tid  = threadIdx.x;
  const int lane = tid & 31;
  const int w    = tid >> 5;
  const int wr   = w >> 1;                    // 0..3
  const int wc   = w & 1;                     // 0..1
  const int rt0  = blockIdx.y * 16 + wr * 4;  // first of 4 row-tiles
  const int ct0  = blockIdx.x * 8  + wc * 4;  // first of 4 col-tiles

  // lane-biased fragment stream bases: record (tile +i, chunk kc) is at
  // base + (i*KC + kc)*32
  const v16bf* Af = (const v16bf*)AF + ((size_t)rt0 * KC) * 32 + lane;
  const v16bf* Bf = (const v16bf*)BF + ((size_t)ct0 * KC) * 32 + lane;

  const v8f vzero = {0.f, 0.f, 0.f, 0.f, 0.f, 0.f, 0.f, 0.f};
  v8f acc[4][4];
#pragma unroll
  for (int i = 0; i < 4; ++i)
#pragma unroll
    for (int j = 0; j < 4; ++j) acc[i][j] = vzero;

  for (int kc = 0; kc < KC; ++kc) {
    // Unconditional speculative prefetch of the next k-chunk
    // (ISA 7.3: invalid/past-end speculative prefetches are dropped).
    __builtin_prefetch(Af + (size_t)(kc + 1) * 32, 0, 0);
    __builtin_prefetch(Bf + (size_t)(kc + 1) * 32, 0, 0);

    v16bf a[4], b[4];
#pragma unroll
    for (int i = 0; i < 4; ++i)
      a[i] = Af[((size_t)i * KC + kc) * 32];
#pragma unroll
    for (int j = 0; j < 4; ++j)
      b[j] = Bf[((size_t)j * KC + kc) * 32];
#pragma unroll
    for (int i = 0; i < 4; ++i)
#pragma unroll
      for (int j = 0; j < 4; ++j)
        acc[i][j] = __builtin_amdgcn_wmma_f32_16x16x32_bf16(
            false, a[i], false, b[j], (short)0, acc[i][j], false, false);
  }

  const int crow = (lane >> 4) * 8;   // C/D: rows crow..crow+7, col = lane&15
  const int ccol = lane & 15;
#pragma unroll
  for (int i = 0; i < 4; ++i)
#pragma unroll
    for (int j = 0; j < 4; ++j) {
#pragma unroll
      for (int vg = 0; vg < 8; ++vg) {
        const int grow = (rt0 + i) * 16 + crow + vg;
        const int gcol = (ct0 + j) * 16 + ccol;
        float vout = acc[i][j][vg];
        if (MODE == 1) {
          vout = vout / (1.0f + expf(-vout));                        // swish
        } else if (MODE == 2) {
          float ls = fminf(vout, 0.0f) - log1pf(expf(-fabsf(vout))); // logsigmoid
          vout = expf(ls * GATE_INV);                                // ge = exp(f)
        }
        C[(size_t)grow * N + gcol] = vout;
      }
    }
}

// ============================================================================
// Recurrence: one workgroup per (b,h); hk[M][K]+hv[M][V] resident in LDS.
// ============================================================================
__global__ __launch_bounds__(256)
void gsa_recurrence(const float* __restrict__ q, const float* __restrict__ k,
                    const float* __restrict__ v, const float* __restrict__ ge,
                    float* __restrict__ o) {
  extern __shared__ float smem[];
  float* hk  = smem;                    // [MS][KH+1]
  float* hv  = hk + MS * (KH + 1);      // [MS][KH]
  float* sq  = hv + MS * KH;            // KH
  float* sk  = sq + KH;                 // KH
  float* sv  = sk + KH;                 // KH
  float* sge = sv + KH;                 // MS
  float* sp  = sge + MS;                // MS (logits -> probs)
  float* spe = sp + MS;                 // MS

  const int tid = threadIdx.x;
  const int b   = blockIdx.x / Hc;
  const int h   = blockIdx.x % Hc;

  for (int i = tid; i < MS * (KH + 1) + MS * KH; i += 256) hk[i] = 0.0f;
  __syncthreads();

  const int m4 = tid >> 2;   // slot owned by this 4-thread group
  const int j4 = tid & 3;    // K-quarter

  for (int t = 0; t < Tc; ++t) {
    const size_t rowQ = (size_t)(b * Tc + t) * (Hc * KH) + (size_t)h * KH;
    const size_t rowG = (size_t)(b * Tc + t) * (Hc * MS) + (size_t)h * MS;
    sq[tid] = q[rowQ + tid];
    sk[tid] = k[rowQ + tid];
    sv[tid] = v[rowQ + tid];
    if (tid < MS) sge[tid] = ge[rowG + tid];
    __syncthreads();

    // hk = hk*ge + k (outer) s;  logits[m] = scale * q . hk[:,m]
    {
      const float gem = sge[m4];
      const float sm  = 1.0f - gem;     // s = 1 - exp(f)
      float* hrow = hk + m4 * (KH + 1);
      float part = 0.0f;
#pragma unroll 4
      for (int i = 0; i < 64; ++i) {
        const int kkk = j4 * 64 + i;
        float hval = hrow[kkk] * gem + sk[kkk] * sm;
        hrow[kkk] = hval;
        part += sq[kkk] * hval;
      }
      part += __shfl_xor(part, 1, 32);
      part += __shfl_xor(part, 2, 32);
      if (j4 == 0) sp[m4] = part * QK_SCALE;
    }
    __syncthreads();

    // softmax over MS slots
    float exv = 0.0f;
    if (tid < MS) {
      float mx = -3.4e38f;
      for (int i = 0; i < MS; ++i) mx = fmaxf(mx, sp[i]);
      exv = expf(sp[tid] - mx);
      spe[tid] = exv;
    }
    __syncthreads();
    if (tid < MS) {
      float sum = 0.0f;
      for (int i = 0; i < MS; ++i) sum += spe[i];
      sp[tid] = exv / sum;
    }
    __syncthreads();

    // hv = hv*ge + s (outer) v;  o = p . hv
    {
      const float vt = sv[tid];
      float oacc = 0.0f;
#pragma unroll 4
      for (int m = 0; m < MS; ++m) {
        const float gem = sge[m];
        float hval = hv[m * KH + tid] * gem + (1.0f - gem) * vt;
        hv[m * KH + tid] = hval;
        oacc += sp[m] * hval;
      }
      o[((size_t)(b * Tc + t) * Hc + h) * KH + tid] = oacc;
    }
    __syncthreads();
  }
}

// ============================================================================
// Per-head RMSNorm; writes bf16 directly in A-fragment order for the Wo GEMM.
// ============================================================================
__global__ __launch_bounds__(256)
void rmsnorm_pack(const float* __restrict__ o, const float* __restrict__ wgt,
                  __bf16* __restrict__ outF) {
  __shared__ float red[8];
  __shared__ float invs;
  const int row = blockIdx.x;           // (b,t,h)
  const int tid = threadIdx.x;          // v
  float val = o[(size_t)row * KH + tid];
  float ss = val * val;
#pragma unroll
  for (int off = 16; off > 0; off >>= 1) ss += __shfl_down(ss, off, 32);
  if ((tid & 31) == 0) red[tid >> 5] = ss;
  __syncthreads();
  if (tid == 0) {
    float tot = 0.0f;
    for (int i = 0; i < 8; ++i) tot += red[i];
    invs = rsqrtf(tot * (1.0f / KH) + 1e-5f);
  }
  __syncthreads();

  const int mr = row >> 2;              // matrix row = b*Tc + t   (Hc == 4)
  const int h  = row & 3;
  const int n  = h * KH + tid;          // contraction index (column of o-mat)
  const int rt = mr >> 4, ar = mr & 15;
  const int kc = n >> 5, ko = n & 31;
  const int lane = ar + ((ko >> 3) & 1) * 16;
  const int e = (ko & 7) + (ko >> 4) * 8;
  constexpr int KC = Dc >> 5;           // 32
  outF[(((size_t)rt * KC + kc) * 32 + lane) * 16 + e] =
      (__bf16)(val * invs * wgt[tid]);
}

// ============================================================================
extern "C" void kernel_launch(void* const* d_in, const int* in_sizes, int n_in,
                              void* d_out, int out_size, void* d_ws, size_t ws_size,
                              hipStream_t stream) {
  (void)in_sizes; (void)n_in; (void)out_size; (void)ws_size;
  const float* x  = (const float*)d_in[0];
  const float* Wq = (const float*)d_in[1];
  const float* Wk = (const float*)d_in[2];
  const float* Wv = (const float*)d_in[3];
  const float* Wf = (const float*)d_in[4];
  const float* gw = (const float*)d_in[5];
  const float* Wo = (const float*)d_in[6];
  float* out = (float*)d_out;

  char* ws = (char*)d_ws;
  size_t off = 0;
  auto take = [&](size_t bytes) -> char* {
    char* p = ws + off;
    off += (bytes + 255) & ~(size_t)255;
    return p;
  };
  __bf16* xF  = (__bf16*)take((size_t)BT * Dc * 2);
  __bf16* wqF = (__bf16*)take((size_t)Dc * Dc * 2);
  __bf16* wkF = (__bf16*)take((size_t)Dc * Dc * 2);
  __bf16* wvF = (__bf16*)take((size_t)Dc * Dc * 2);
  __bf16* wfF = (__bf16*)take((size_t)Dc * HM * 2);
  __bf16* woF = (__bf16*)take((size_t)Dc * Dc * 2);
  float*  qb  = (float*)take((size_t)BT * Dc * 4);
  float*  kb  = (float*)take((size_t)BT * Dc * 4);
  float*  vb  = (float*)take((size_t)BT * Dc * 4);
  float*  geb = (float*)take((size_t)BT * HM * 4);
  float*  ob  = (float*)take((size_t)BT * Dc * 4);
  __bf16* onF = (__bf16*)take((size_t)BT * Dc * 2);

  pack_a_frag<<<2048, 256, 0, stream>>>(x, xF, BT, Dc);
  pack_b_frag<<<2048, 256, 0, stream>>>(Wq, wqF, Dc, Dc);
  pack_b_frag<<<2048, 256, 0, stream>>>(Wk, wkF, Dc, Dc);
  pack_b_frag<<<2048, 256, 0, stream>>>(Wv, wvF, Dc, Dc);
  pack_b_frag<<<512,  256, 0, stream>>>(Wf, wfF, Dc, HM);
  pack_b_frag<<<2048, 256, 0, stream>>>(Wo, woF, Dc, Dc);

  dim3 gP(Dc / 128, BT / 256);   // (8, 32)
  gemm_frag_wmma<1><<<gP, 256, 0, stream>>>(xF, wqF, qb, Dc, Dc);   // q = swish
  gemm_frag_wmma<1><<<gP, 256, 0, stream>>>(xF, wkF, kb, Dc, Dc);   // k = swish
  gemm_frag_wmma<0><<<gP, 256, 0, stream>>>(xF, wvF, vb, Dc, Dc);   // v
  dim3 gF(HM / 128, BT / 256);   // (2, 32)
  gemm_frag_wmma<2><<<gF, 256, 0, stream>>>(xF, wfF, geb, HM, Dc);  // ge

  constexpr size_t REC_SMEM =
      (size_t)(MS * (KH + 1) + MS * KH + 3 * KH + 3 * MS) * sizeof(float);
  hipFuncSetAttribute((const void*)gsa_recurrence,
                      hipFuncAttributeMaxDynamicSharedMemorySize, (int)REC_SMEM);
  gsa_recurrence<<<dim3(Bc * Hc), 256, REC_SMEM, stream>>>(qb, kb, vb, geb, ob);

  rmsnorm_pack<<<dim3(BT * Hc), 256, 0, stream>>>(ob, gw, onF);
  gemm_frag_wmma<0><<<gP, 256, 0, stream>>>(onF, woF, out, Dc, Dc); // @ Wo
}
